// MultiNet_70377334112868
// MI455X (gfx1250) — matrix operand order repaired and checked by reading, not compiled, not run on hardware
//
#include <hip/hip_runtime.h>
#include <hip/hip_bf16.h>
#include <cstddef>

// ---------------------------------------------------------------------------
// MultiNet forward for MI455X (gfx1250, wave32, WMMA).
// Dense matmuls: global -> LDS staging -> v_wmma_f32_16x16x32_f16 (f32 acc).
// Irregular graph ops (uniform-segment softmax, edge scatter-mean) are VALU.
// ---------------------------------------------------------------------------

typedef __attribute__((ext_vector_type(16))) _Float16 v16h;
typedef __attribute__((ext_vector_type(8)))  _Float16 v8h;
typedef __attribute__((ext_vector_type(8)))  float    v8f;

#define DEVFN static __device__ __forceinline__

DEVFN float celu1(float v) { return v > 0.f ? v : (__expf(v) - 1.f); }
DEVFN float sigf(float v)  { return 1.f / (1.f + __expf(-v)); }

// ---------------------------------------------------------------------------
// WMMA GEMM:  Y[M,N] = act( X[M,K] @ W^T + bias )
//   wt==0 : W stored [N,K] row-major (torch Linear weight)
//   wt==1 : W stored [K,N] row-major (used for x @ root)
//   act==1: CELU(alpha=1)
// One wave (block of 32) computes one 16x16 output tile.
// Tiles are staged through LDS as f16 (zero-padded to K32), then fragments
// are built with unconditional 16B-aligned LDS vector reads.
// Operand packing per CDNA5 ISA 7.12.2 (16-bit A 16x32, B 32x16, f32 C/D).
// ---------------------------------------------------------------------------
template <int K>
__global__ void gemm_wmma_kernel(const float* __restrict__ X,
                                 const float* __restrict__ W,
                                 const float* __restrict__ bias,
                                 float* __restrict__ Y,
                                 int M, int N, int act, int wt)
{
    constexpr int K32 = (K + 31) & ~31;
    __shared__ alignas(16) _Float16 As[16 * K32];
    __shared__ alignas(16) _Float16 Bs[16 * K32];

    int tilesN = (N + 15) >> 4;
    int tile = blockIdx.x;
    int tm = tile / tilesN;
    int tn = tile - tm * tilesN;
    int lane = threadIdx.x;          // 0..31, one wave per block

    // ---- stage A tile: X rows tm*16..tm*16+15, cols 0..K-1, pad to K32 ----
    const float* Xt = X + (size_t)(tm << 4) * K;
    for (int t = lane; t < 16 * K32; t += 32) {
        int r = t / K32, k = t - r * K32;           // K32 const -> strength-reduced
        float v = (k < K) ? Xt[(size_t)r * K + k] : 0.f;
        As[t] = (_Float16)v;
    }
    // ---- stage B tile: output cols tn*16..tn*16+15 (rows of W if wt==0) ----
    if (wt == 0) {
        for (int t = lane; t < 16 * K32; t += 32) {
            int n = t / K32, k = t - n * K32;
            int col = (tn << 4) + n;
            float v = (col < N && k < K) ? W[(size_t)col * K + k] : 0.f;
            Bs[t] = (_Float16)v;
        }
    } else {
        for (int t = lane; t < 16 * K32; t += 32) {
            int n = t / K32, k = t - n * K32;
            int col = (tn << 4) + n;
            float v = (col < N && k < K) ? W[(size_t)k * N + col] : 0.f;
            Bs[t] = (_Float16)v;
        }
    }
    __syncthreads();   // single-wave workgroup: S_NOP, but orders DS store->load

    int h = lane >> 4;               // half-wave
    int m = lane & 15;

    v8f acc = {};
#pragma unroll
    for (int k0 = 0; k0 < K32; k0 += 32) {
        // A fragment: lane (h,m) needs K = {8h..8h+7} and {16+8h..16+8h+7}
        const v8h* arow = (const v8h*)(As + m * K32 + k0);
        v8h a0 = arow[h];
        v8h a1 = arow[2 + h];
        // B fragment: lane (h,m) needs K = {8h..8h+7} and {16+8h..16+8h+7} of output column m
        const v8h* brow = (const v8h*)(Bs + m * K32 + k0);
        v8h b0 = brow[h];
        v8h b1 = brow[2 + h];
        v16h a, b;
#pragma unroll
        for (int j = 0; j < 8; ++j) {
            a[j] = a0[j]; a[8 + j] = a1[j];
            b[j] = b0[j]; b[8 + j] = b1[j];
        }
        acc = __builtin_amdgcn_wmma_f32_16x16x32_f16(
                  false, a, false, b, (short)0, acc, false, false);
    }

    int col = (tn << 4) + m;
    if (col < N) {
        float bb = bias ? bias[col] : 0.f;
#pragma unroll
        for (int r = 0; r < 8; ++r) {
            int row = (tm << 4) + r + 8 * h;   // D: VGPR r, half h -> M = r + 8h
            float v = acc[r] + bb;
            if (act == 1) v = celu1(v);
            Y[(size_t)row * N + col] = v;
        }
    }
}

// ---------------------------------------------------------------------------
// Elementwise / graph kernels
// ---------------------------------------------------------------------------
__global__ void fill_kernel(float* p, float v, int n)
{
    int i = blockIdx.x * blockDim.x + threadIdx.x;
    if (i < n) p[i] = v;
}

// GRU combine (torch gate order r,z,n): out = (1-z)*n + z*h
__global__ void gru_kernel(const float* __restrict__ gi, const float* __restrict__ gh,
                           const float* __restrict__ hprev,
                           float* __restrict__ out1, float* __restrict__ out2,
                           int n, int d)
{
    int idx = blockIdx.x * blockDim.x + threadIdx.x;
    if (idx >= n * d) return;
    int s = idx / d, k = idx - s * d;
    const float* Gi = gi + (size_t)s * 3 * d;
    const float* Gh = gh + (size_t)s * 3 * d;
    float r  = sigf(Gi[k]         + Gh[k]);
    float z  = sigf(Gi[d + k]     + Gh[d + k]);
    float nn = tanhf(Gi[2 * d + k] + r * Gh[2 * d + k]);
    float hv = (1.f - z) * nn + z * hprev[idx];
    out1[idx] = hv;
    if (out2) out2[idx] = hv;
}

// LSTM combine (torch gate order i,f,g,o); also writes h into q_star[:, 0:d]
__global__ void lstm_kernel(const float* __restrict__ gi, const float* __restrict__ gh,
                            float* __restrict__ h, float* __restrict__ c,
                            float* __restrict__ qstar, int n, int d)
{
    int idx = blockIdx.x * blockDim.x + threadIdx.x;
    if (idx >= n * d) return;
    int s = idx / d, k = idx - s * d;
    const float* Gi = gi + (size_t)s * 4 * d;
    const float* Gh = gh + (size_t)s * 4 * d;
    float ig = Gi[k]         + Gh[k];
    float fg = Gi[d + k]     + Gh[d + k];
    float gg = Gi[2 * d + k] + Gh[2 * d + k];
    float og = Gi[3 * d + k] + Gh[3 * d + k];
    float cv = sigf(fg) * c[idx] + sigf(ig) * tanhf(gg);
    float hv = sigf(og) * tanhf(cv);
    c[idx] = cv;
    h[idx] = hv;
    qstar[(size_t)s * 2 * d + k] = hv;
}

// e[i] = dot(x[i,:], h[i/segsz,:])   (uniform contiguous segments)
__global__ void dot_kernel(const float* __restrict__ x, const float* __restrict__ h,
                           float* __restrict__ e, int items, int d, int segsz)
{
    int i = blockIdx.x * blockDim.x + threadIdx.x;
    if (i >= items) return;
    int s = i / segsz;
    float acc = 0.f;
    for (int k = 0; k < d; ++k)
        acc += x[(size_t)i * d + k] * h[(size_t)s * d + k];
    e[i] = acc;
}

// Segment softmax + weighted sum -> q_star[:, d:2d]  (uniform segments)
__global__ void attend_kernel(const float* __restrict__ x, const float* __restrict__ e,
                              float* __restrict__ qstar, int nseg, int d, int segsz)
{
    int idx = blockIdx.x * blockDim.x + threadIdx.x;
    if (idx >= nseg * d) return;
    int s = idx / d, dim = idx - s * d;
    int base = s * segsz;
    float mx = -1e30f;
    for (int j = 0; j < segsz; ++j) mx = fmaxf(mx, e[base + j]);
    float sum = 0.f, r = 0.f;
    for (int j = 0; j < segsz; ++j) {
        float w = __expf(e[base + j] - mx);
        sum += w;
        r   += w * x[(size_t)(base + j) * d + dim];
    }
    qstar[(size_t)s * 2 * d + d + dim] = r / sum;
}

// NNConv message: msg[e,o] = sum_i x[src[e],i] * W[e,i,o]; scatter-add to dst
__global__ void nnconv_msg_kernel(const float* __restrict__ x, const float* __restrict__ W,
                                  const int* __restrict__ src, const int* __restrict__ dst,
                                  float* __restrict__ sbuf, int E, int d)
{
    int idx = blockIdx.x * blockDim.x + threadIdx.x;
    if (idx >= E * d) return;
    int e = idx / d, o = idx - e * d;
    const float* We = W + (size_t)e * d * d;
    const float* xs = x + (size_t)src[e] * d;
    float acc = 0.f;
    for (int i = 0; i < d; ++i) acc += xs[i] * We[(size_t)i * d + o];
    atomicAdd(&sbuf[(size_t)dst[e] * d + o], acc);
}

__global__ void count_kernel(const int* __restrict__ dst, float* __restrict__ cnt, int E)
{
    int e = blockIdx.x * blockDim.x + threadIdx.x;
    if (e < E) atomicAdd(&cnt[dst[e]], 1.f);
}

// out = celu( sbuf/max(cnt,1) + rootpart )   (rootpart already has conv bias)
__global__ void conv_combine_kernel(const float* __restrict__ sbuf, const float* __restrict__ cnt,
                                    const float* __restrict__ rootp, float* __restrict__ out,
                                    int n, int d)
{
    int idx = blockIdx.x * blockDim.x + threadIdx.x;
    if (idx >= n * d) return;
    float c = fmaxf(cnt[idx / d], 1.f);
    out[idx] = celu1(sbuf[idx] / c + rootp[idx]);
}

__global__ void gather_kernel(const float* __restrict__ in, const int* __restrict__ seg,
                              float* __restrict__ out, int items, int d)
{
    int idx = blockIdx.x * blockDim.x + threadIdx.x;
    if (idx >= items * d) return;
    int i = idx / d, k = idx - i * d;
    out[idx] = in[(size_t)seg[i] * d + k];
}

__global__ void concat_kernel(const float* __restrict__ a, const float* __restrict__ b,
                              float* __restrict__ out, int n, int da, int db)
{
    int idx = blockIdx.x * blockDim.x + threadIdx.x;
    int dt = da + db;
    if (idx >= n * dt) return;
    int r = idx / dt, c = idx - r * dt;
    out[idx] = (c < da) ? a[(size_t)r * da + c] : b[(size_t)r * db + (c - da)];
}

// ---------------------------------------------------------------------------
// Host-side orchestration
// ---------------------------------------------------------------------------
namespace {

constexpr int Bg = 128;
constexpr int N_ATOM = 2048, E_ATOM = 24576;
constexpr int N_ORB = 16384, E_ORB = 131072;
constexpr int DA = 32, EA = 16;
constexpr int DO = 16, EO = 16;
constexpr int MP_STEP = 6, S2S_STEP = 6;

inline int ew_blocks(int n) { return (n + 255) / 256; }

// K is restricted to the shapes appearing in this network.
void gemm(const float* X, const float* W, const float* b, float* Y,
          int M, int N, int K, int act, int wt, hipStream_t st)
{
    int tiles = (M >> 4) * ((N + 15) >> 4);
    switch (K) {
    case 8:
        gemm_wmma_kernel<8><<<tiles, 32, 0, st>>>(X, W, b, Y, M, N, act, wt); break;
    case 16:
        gemm_wmma_kernel<16><<<tiles, 32, 0, st>>>(X, W, b, Y, M, N, act, wt); break;
    case 32:
        gemm_wmma_kernel<32><<<tiles, 32, 0, st>>>(X, W, b, Y, M, N, act, wt); break;
    case 48:
        gemm_wmma_kernel<48><<<tiles, 32, 0, st>>>(X, W, b, Y, M, N, act, wt); break;
    case 64:
        gemm_wmma_kernel<64><<<tiles, 32, 0, st>>>(X, W, b, Y, M, N, act, wt); break;
    case 192:
        gemm_wmma_kernel<192><<<tiles, 32, 0, st>>>(X, W, b, Y, M, N, act, wt); break;
    default: break; // not used
    }
}

void fill(float* p, float v, int n, hipStream_t st)
{
    fill_kernel<<<ew_blocks(n), 256, 0, st>>>(p, v, n);
}

// Set2Set: x [n*segsz, d] -> q_star [n, 2d]
void set2set(const float* x, int n, int d, int segsz,
             const float* wih, const float* whh, const float* bih, const float* bhh,
             float* h, float* c, float* q, float* gi, float* gh, float* e,
             hipStream_t st)
{
    fill(h, 0.f, n * d, st);
    fill(c, 0.f, n * d, st);
    fill(q, 0.f, n * 2 * d, st);
    int items = n * segsz;
    for (int s = 0; s < S2S_STEP; ++s) {
        gemm(q, wih, bih, gi, n, 4 * d, 2 * d, 0, 0, st);   // gi = q_star @ Wih^T + bih
        gemm(h, whh, bhh, gh, n, 4 * d, d,     0, 0, st);   // gh = h @ Whh^T + bhh
        lstm_kernel<<<ew_blocks(n * d), 256, 0, st>>>(gi, gh, h, c, q, n, d);
        dot_kernel<<<ew_blocks(items), 256, 0, st>>>(x, h, e, items, d, segsz);
        attend_kernel<<<ew_blocks(n * d), 256, 0, st>>>(x, e, q, n, d, segsz);
    }
}

void gru(const float* xin, const float* hprev, float* out1, float* out2,
         const float* wih, const float* whh, const float* bih, const float* bhh,
         float* gi, float* gh, int n, int din, int d, hipStream_t st)
{
    gemm(xin,   wih, bih, gi, n, 3 * d, din, 0, 0, st);
    gemm(hprev, whh, bhh, gh, n, 3 * d, d,   0, 0, st);
    gru_kernel<<<ew_blocks(n * d), 256, 0, st>>>(gi, gh, hprev, out1, out2, n, d);
}

} // namespace

extern "C" void kernel_launch(void* const* d_in, const int* in_sizes, int n_in,
                              void* d_out, int out_size, void* d_ws, size_t ws_size,
                              hipStream_t stream)
{
    (void)in_sizes; (void)n_in; (void)out_size; (void)ws_size;

    const float* atom_x        = (const float*)d_in[0];   // [2048,16]
    const float* atom_eattr    = (const float*)d_in[1];   // [24576,8]
    const int*   atom_eidx     = (const int*)  d_in[2];   // [2,24576]
    const float* orb_x         = (const float*)d_in[4];   // [16384,16]
    const float* orb_eattr     = (const float*)d_in[5];   // [131072,8]
    const int*   orb_eidx      = (const int*)  d_in[6];   // [2,131072]
    const int*   orb_bidx      = (const int*)  d_in[8];   // [16384] orbital->atom

    // Params: JAX tree-flatten order (sorted dict keys, tuple order, b<w).
    const float* P[68];
    for (int i = 0; i < 68; ++i) P[i] = (const float*)d_in[9 + i];
    // 0-3   atom_conv{bias, nn.b, nn.w, root}
    // 4-7   atom_gru{bhh,bih,whh,wih}
    // 8-11  atom_lin0{0.b,0.w,1.b,1.w}      12-15 atom_lin1
    // 16-17 atom_lin2{b,w}                   18-21 atom_s2s{bhh,bih,whh,wih}
    // 22-25 cross_a2o_gru                    26-29 cross_a2o_lin
    // 30-33 cross_lin0{0.b,0.w,1.b,1.w}      34-37 cross_o2a_gru
    // 38-41 cross_o2a_lin                    42-45 cross_o2a_s2s
    // 46-49 orbital_conv                     50-53 orbital_gru
    // 54-57 orbital_lin0                     58-61 orbital_lin1
    // 62-63 orbital_lin2                     64-67 orbital_s2s

    const int* asrc = atom_eidx;           const int* adst = atom_eidx + E_ATOM;
    const int* osrc = orb_eidx;            const int* odst = orb_eidx + E_ORB;

    // -------------------- workspace (bump allocator, floats) ----------------
    float* base = (float*)d_ws;
    size_t off = 0;
    auto alloc = [&](size_t n) { float* p = base + off; off += (n + 63) & ~size_t(63); return p; };

    float* atom_v = alloc((size_t)N_ATOM * DA);
    float* orb_v  = alloc((size_t)N_ORB * DO);
    float* atom_e = alloc((size_t)E_ATOM * EA);
    float* orb_e  = alloc((size_t)E_ORB * EO);
    float* a_h    = alloc((size_t)N_ATOM * DA);
    float* o_h    = alloc((size_t)N_ORB * DO);
    float* a2o_h  = alloc((size_t)N_ORB * DO);
    float* o2a_h  = alloc((size_t)N_ATOM * DA);
    float* Watom  = alloc((size_t)E_ATOM * DA * DA);   // 24576 x 1024
    float* Worb   = alloc((size_t)E_ORB * DO * DO);    // 131072 x 256
    float* cntA   = alloc(N_ATOM);
    float* cntO   = alloc(N_ORB);
    float* sA     = alloc((size_t)N_ATOM * DA);
    float* sO     = alloc((size_t)N_ORB * DO);
    float* tA     = alloc((size_t)E_ORB * 16);         // general temp (big)
    float* tB     = alloc((size_t)E_ORB * 16);
    float* gi     = alloc((size_t)N_ORB * 48);
    float* gh     = alloc((size_t)N_ORB * 48);
    float* hS     = alloc((size_t)N_ATOM * DA);
    float* cS     = alloc((size_t)N_ATOM * DA);
    float* qS     = alloc((size_t)N_ATOM * 2 * DA);
    float* xS     = alloc((size_t)N_ORB * DO);
    float* eB     = alloc(N_ORB);
    float* m_o    = alloc((size_t)N_ORB * DO);
    float* a_out  = alloc((size_t)Bg * 4 * DA);
    float* o_out  = alloc((size_t)Bg * 4 * DO);
    float* xcat   = alloc((size_t)Bg * (4 * DA + 4 * DO));
    float* y48    = alloc((size_t)Bg * 48);

    hipStream_t st = stream;

    // -------------------- input embeddings (mlp2 = lin-celu-lin-celu) ------
    gemm(atom_x, P[9],  P[8],  tA,     N_ATOM, 32, 16, 1, 0, st);
    gemm(tA,     P[11], P[10], atom_v, N_ATOM, DA, 32, 1, 0, st);
    gemm(atom_eattr, P[13], P[12], tA,     E_ATOM, 16, 8,  1, 0, st);
    gemm(tA,         P[15], P[14], atom_e, E_ATOM, EA, 16, 1, 0, st);
    gemm(orb_x, P[55], P[54], tA,    N_ORB, 32, 16, 1, 0, st);
    gemm(tA,    P[57], P[56], orb_v, N_ORB, DO, 32, 1, 0, st);
    gemm(orb_eattr, P[59], P[58], tA,    E_ORB, 16, 8,  1, 0, st);
    gemm(tA,        P[61], P[60], orb_e, E_ORB, EO, 16, 1, 0, st);

    // hidden states start as the embeddings
    hipMemcpyAsync(a_h,   atom_v, sizeof(float) * N_ATOM * DA, hipMemcpyDeviceToDevice, st);
    hipMemcpyAsync(o2a_h, atom_v, sizeof(float) * N_ATOM * DA, hipMemcpyDeviceToDevice, st);
    hipMemcpyAsync(o_h,   orb_v,  sizeof(float) * N_ORB * DO,  hipMemcpyDeviceToDevice, st);
    hipMemcpyAsync(a2o_h, orb_v,  sizeof(float) * N_ORB * DO,  hipMemcpyDeviceToDevice, st);

    // -------------------- loop-invariant precompute -------------------------
    // Per-edge NNConv weights (edge MLP output is constant across MP steps).
    gemm(atom_e, P[2],  P[1],  Watom, E_ATOM, DA * DA, EA, 0, 0, st);
    gemm(orb_e,  P[48], P[47], Worb,  E_ORB,  DO * DO, EO, 0, 0, st);
    // In-degree counts (constant).
    fill(cntA, 0.f, N_ATOM, st);
    fill(cntO, 0.f, N_ORB, st);
    count_kernel<<<ew_blocks(E_ATOM), 256, 0, st>>>(adst, cntA, E_ATOM);
    count_kernel<<<ew_blocks(E_ORB),  256, 0, st>>>(odst, cntO, E_ORB);

    // -------------------- message passing -----------------------------------
    for (int step = 0; step < MP_STEP; ++step) {
        // orbital -> atom: Set2Set over 8 orbitals per atom
        gemm(orb_v, P[39], P[38], tA, N_ORB, 32, DO, 1, 0, st);
        gemm(tA,    P[41], P[40], xS, N_ORB, 16, 32, 1, 0, st);
        set2set(xS, N_ATOM, 16, 8, P[45], P[44], P[43], P[42],
                hS, cS, qS, gi, gh, eB, st);                       // qS [2048,32]
        gru(qS, o2a_h, atom_v, o2a_h, P[37], P[36], P[35], P[34],
            gi, gh, N_ATOM, DA, DA, st);

        // atom -> orbital: gather atom message per orbital
        gemm(atom_v, P[27], P[26], tA, N_ATOM, 64, DA, 1, 0, st);
        gemm(tA,     P[29], P[28], tB, N_ATOM, DO, 64, 1, 0, st);
        gather_kernel<<<ew_blocks(N_ORB * DO), 256, 0, st>>>(tB, orb_bidx, m_o, N_ORB, DO);
        gru(m_o, a2o_h, orb_v, a2o_h, P[25], P[24], P[23], P[22],
            gi, gh, N_ORB, DO, DO, st);

        // atom graph NNConv(mean) + CELU + GRU
        fill(sA, 0.f, N_ATOM * DA, st);
        nnconv_msg_kernel<<<ew_blocks(E_ATOM * DA), 256, 0, st>>>(atom_v, Watom, asrc, adst, sA, E_ATOM, DA);
        gemm(atom_v, P[3], P[0], tB, N_ATOM, DA, DA, 0, 1, st);    // x @ root + bias
        conv_combine_kernel<<<ew_blocks(N_ATOM * DA), 256, 0, st>>>(sA, cntA, tB, tA, N_ATOM, DA);
        gru(tA, a_h, atom_v, a_h, P[7], P[6], P[5], P[4],
            gi, gh, N_ATOM, DA, DA, st);

        // orbital graph NNConv(mean) + CELU + GRU
        fill(sO, 0.f, N_ORB * DO, st);
        nnconv_msg_kernel<<<ew_blocks(E_ORB * DO), 256, 0, st>>>(orb_v, Worb, osrc, odst, sO, E_ORB, DO);
        gemm(orb_v, P[49], P[46], tB, N_ORB, DO, DO, 0, 1, st);
        conv_combine_kernel<<<ew_blocks(N_ORB * DO), 256, 0, st>>>(sO, cntO, tB, tA, N_ORB, DO);
        gru(tA, o_h, orb_v, o_h, P[53], P[52], P[51], P[50],
            gi, gh, N_ORB, DO, DO, st);
    }

    // -------------------- readout -------------------------------------------
    // atom Set2Set over 16 atoms per graph -> [128,64]
    set2set(atom_v, Bg, DA, N_ATOM / Bg, P[21], P[20], P[19], P[18],
            hS, cS, qS, gi, gh, eB, st);
    gemm(qS, P[17], P[16], a_out, Bg, 4 * DA, 2 * DA, 1, 0, st);   // [128,128]

    // orbital Set2Set over 128 orbitals per graph -> [128,32]
    set2set(orb_v, Bg, DO, N_ORB / Bg, P[67], P[66], P[65], P[64],
            hS, cS, qS, gi, gh, eB, st);
    gemm(qS, P[63], P[62], o_out, Bg, 4 * DO, 2 * DO, 1, 0, st);   // [128,64]

    concat_kernel<<<ew_blocks(Bg * 192), 256, 0, st>>>(a_out, o_out, xcat, Bg, 4 * DA, 4 * DO);
    gemm(xcat, P[31], P[30], y48, Bg, 48, 192, 1, 0, st);
    gemm(y48,  P[33], P[32], (float*)d_out, Bg, 12, 48, 0, 0, st); // [128,12]
}
